// WordEncoder_43920335569469
// MI455X (gfx1250) — matrix operand order repaired and checked by reading, not compiled, not run on hardware
//
#include <hip/hip_runtime.h>
#include <stdint.h>

// ---------------- problem constants ----------------
#define E_   128
#define U_   256
#define D_   256
#define B_   128
#define T_   1024
#define NG   4          // gates i,f,g,o
#define NWG_REC 16      // persistent workgroups in recurrent kernel
#define NCHUNK 12       // K chunks per step: 4 (x@Wk, K=128) + 8 (h@Wr, K=256)

typedef __attribute__((ext_vector_type(16))) __bf16 v16bf;
typedef __attribute__((ext_vector_type(8)))  float  v8f;

union ABReg { uint4 q[2]; v16bf v; };

__device__ __forceinline__ unsigned short f2bf(float f) {
  unsigned u = __float_as_uint(f);
  u += 0x7FFFu + ((u >> 16) & 1u);       // round-to-nearest-even
  return (unsigned short)(u >> 16);
}

// A operand 16x32 bf16: lane holds row m=lane&15;
// lanes 0-15: K={0..7,16..23}(+kk)  lanes 16-31: K={8..15,24..31}(+kk)
// p already includes  m*stride + kk + half*8
template <typename P>
__device__ __forceinline__ v16bf ldA(const P* p) {
  ABReg r;
  r.q[0] = *(const uint4*)(p);
  r.q[1] = *(const uint4*)(p + 16);
  return r.v;
}
// B operand 32x16 bf16 from [n][K] storage: lane holds col n=lane&15;
// lanes 0-15: K={0..7,16..23}(+kk)  lanes 16-31: K={8..15,24..31}(+kk)
// p already includes  n*stride + kk + half*8
template <typename P>
__device__ __forceinline__ v16bf ldB(const P* p) {
  ABReg r;
  r.q[0] = *(const uint4*)(p);
  r.q[1] = *(const uint4*)(p + 16);
  return r.v;
}

__device__ __forceinline__ float sigf(float x)     { return 1.0f / (1.0f + __expf(-x)); }
__device__ __forceinline__ float tanhfast(float x) { return 2.0f / (1.0f + __expf(-2.0f * x)) - 1.0f; }

// ---------------- kernel 1: embedding gather + mask ----------------
__global__ __launch_bounds__(128) void gather_embed(
    const int* __restrict__ tokens, const float* __restrict__ emb,
    unsigned short* __restrict__ Xb, float* __restrict__ maskT) {
  const int row = blockIdx.x;            // row = t*B + b
  const int t = row >> 7;
  const int bb = row & (B_ - 1);
  const int tok = tokens[bb * T_ + t];
  const int e = threadIdx.x;
  Xb[row * E_ + e] = f2bf(emb[tok * E_ + e]);
  if (e == 0) maskT[row] = (tok != 0) ? 1.0f : 0.0f;
}

// ---------------- kernel 2: weight transpose + bf16 convert ----------------
__global__ __launch_bounds__(256) void prep_weights(
    const float* __restrict__ Wk, const float* __restrict__ Wr, const float* __restrict__ Wd,
    unsigned short* __restrict__ WkT, unsigned short* __restrict__ WrT,
    unsigned short* __restrict__ WdT) {
  const int i = blockIdx.x * 256 + threadIdx.x;
  const int NK = NG * U_ * E_;   // 131072
  const int NR = NG * U_ * U_;   // 262144
  const int ND = D_ * U_;        // 65536
  if (i < NK) {                                    // WkT[n][e] <- Wk[e][n]
    const int n = i >> 7, e = i & 127;
    WkT[i] = f2bf(Wk[e * (NG * U_) + n]);
  } else if (i < NK + NR) {                        // WrT[n][k] <- Wr[k][n]
    const int j = i - NK;
    const int n = j >> 8, k = j & 255;
    WrT[j] = f2bf(Wr[k * (NG * U_) + n]);
  } else if (i < NK + NR + ND) {                   // WdT[n][k] <- Wd[k][n]
    const int j = i - NK - NR;
    const int n = j >> 8, k = j & 255;
    WdT[j] = f2bf(Wd[k * D_ + n]);
  }
}

// ---------------- kernel 3: persistent masked-LSTM recurrence ----------------
// Wave (of 8 per WG): gate g = wave&3, batch-half mh = wave>>2.
// Weight slice resident in 96 VGPRs for all 1024 steps; K loop is pure
// global A loads (double-buffered, schedule pinned with sched_barrier) +
// register-B WMMA chains. Gates exchanged via 32 KB LDS for pointwise phase.
__global__ __launch_bounds__(256) __attribute__((amdgpu_waves_per_eu(2)))
void lstm_rec(
    const unsigned short* __restrict__ Xb,    // [T][B][E] bf16
    const unsigned short* __restrict__ WkT,   // [4U][E]  bf16
    const unsigned short* __restrict__ WrT,   // [4U][U]  bf16
    const float* __restrict__ bias,           // [4U]
    const float* __restrict__ maskT,          // [T][B]
    unsigned short* __restrict__ hbuf,        // 2 x [B][U] bf16 (double buffer)
    unsigned* __restrict__ barrier_cnt,
    float* __restrict__ out)                  // [B][T][D] — receives h_t
{
  __shared__ __align__(16) float zbuf[NG][8][32][8];   // 32 KB gate exchange

  const int wg   = blockIdx.x;     // u-tile (0..15)
  const int tid  = threadIdx.x;
  const int lane = tid & 31;
  const int wave = tid >> 5;
  const int g    = wave & 3;       // gate handled in K loop
  const int mh   = wave >> 2;      // batch half (m-tiles mh*4 .. mh*4+3)
  const int ubase = wg * 16;
  const int half = lane >> 4;
  const int n    = lane & 15;      // A row within tile / B col / C col
  const int ucol = ubase + n;

  // ---- loop-invariant weight tiles -> registers (96 VGPRs) ----
  v16bf Breg[NCHUNK];
  {
    const int col = g * U_ + ubase + n;          // global gate column
    #pragma unroll
    for (int c = 0; c < 4; ++c)
      Breg[c] = ldB(WkT + col * E_ + c * 32 + half * 8);
    #pragma unroll
    for (int c = 0; c < 8; ++c)
      Breg[4 + c] = ldB(WrT + col * U_ + c * 32 + half * 8);
  }
  const float bg = bias[g * U_ + ucol];          // bias for K-loop gate
  // A-operand per-lane offsets for the 4 m-tiles this wave multiplies
  int xoffj[4], hoffj[4];
  #pragma unroll
  for (int j = 0; j < 4; ++j) {
    const int row = (mh * 4 + j) * 16 + n;
    xoffj[j] = row * E_ + half * 8;
    hoffj[j] = row * U_ + half * 8;
  }

  // pointwise-phase state: wave owns m-tile 'wave' there
  v8f creg, hreg;
  #pragma unroll
  for (int k = 0; k < 8; ++k) { creg[k] = 0.0f; hreg[k] = 0.0f; }

  // preload chunk-0 (x) A tiles for t=0
  v16bf aC[4], aN[4];
  #pragma unroll
  for (int j = 0; j < 4; ++j) aC[j] = ldA(Xb + xoffj[j]);

  __syncthreads();   // zbuf ready-for-use alignment across waves

  for (int t = 0; t < T_; ++t) {
    const int cur = t & 1, nxt = cur ^ 1;
    const unsigned short* xrow = Xb + (t * B_) * E_;
    const unsigned short* hrow = hbuf + cur * (B_ * U_);

    v8f acc[4];
    #pragma unroll
    for (int j = 0; j < 4; ++j) {
      #pragma unroll
      for (int k = 0; k < 8; ++k) acc[j][k] = bg;
    }

    // ---- K loop: chunk-major, double-buffered A, schedule pinned ----
    #pragma unroll
    for (int c = 0; c < NCHUNK; ++c) {
      if (c + 1 < NCHUNK) {                    // issue chunk c+1's 8 loads
        const int cn = c + 1;
        #pragma unroll
        for (int j = 0; j < 4; ++j)
          aN[j] = (cn < 4) ? ldA(xrow + xoffj[j] + cn * 32)
                           : ldA(hrow + hoffj[j] + (cn - 4) * 32);
      }
      __builtin_amdgcn_sched_barrier(0);       // keep loads above the WMMAs
      #pragma unroll
      for (int j = 0; j < 4; ++j)
        acc[j] = __builtin_amdgcn_wmma_f32_16x16x32_bf16(
            false, aC[j], false, Breg[c], (short)0, acc[j], false, false);
      __builtin_amdgcn_sched_barrier(0);       // keep next loads below prev WMMAs
      #pragma unroll
      for (int j = 0; j < 4; ++j) aC[j] = aN[j];
    }

    // ---- gate exchange through LDS ----
    #pragma unroll
    for (int j = 0; j < 4; ++j) {
      float* zp = &zbuf[g][mh * 4 + j][lane][0];
      *(float4*)(zp)     = make_float4(acc[j][0], acc[j][1], acc[j][2], acc[j][3]);
      *(float4*)(zp + 4) = make_float4(acc[j][4], acc[j][5], acc[j][6], acc[j][7]);
    }
    __syncthreads();

    // ---- pointwise: wave owns m-tile 'wave' ----
    v8f z[NG];
    #pragma unroll
    for (int gg = 0; gg < NG; ++gg) {
      const float* zp = &zbuf[gg][wave][lane][0];
      const float4 lo = *(const float4*)(zp);
      const float4 hi = *(const float4*)(zp + 4);
      z[gg][0] = lo.x; z[gg][1] = lo.y; z[gg][2] = lo.z; z[gg][3] = lo.w;
      z[gg][4] = hi.x; z[gg][5] = hi.y; z[gg][6] = hi.z; z[gg][7] = hi.w;
    }

    const float* mrow = maskT + t * B_ + wave * 16 + half * 8;
    const float4 m0 = *(const float4*)(mrow);
    const float4 m1 = *(const float4*)(mrow + 4);
    const float mk[8] = {m0.x, m0.y, m0.z, m0.w, m1.x, m1.y, m1.z, m1.w};

    #pragma unroll
    for (int k = 0; k < 8; ++k) {
      const float iv = sigf(z[0][k]);
      const float fv = sigf(z[1][k]);
      const float gv = tanhfast(z[2][k]);
      const float ov = sigf(z[3][k]);
      const float cn = fv * creg[k] + iv * gv;
      const float hn = ov * tanhfast(cn);
      const bool upd = mk[k] > 0.5f;
      creg[k] = upd ? cn : creg[k];
      hreg[k] = upd ? hn : hreg[k];
    }

    // write h: f32 to output tensor, bf16 to next-step buffer
    unsigned short* hout = hbuf + nxt * (B_ * U_);
    #pragma unroll
    for (int k = 0; k < 8; ++k) {
      const int bb = wave * 16 + half * 8 + k;
      out[(bb * T_ + t) * D_ + ucol] = hreg[k];
      hout[bb * U_ + ucol] = f2bf(hreg[k]);
    }

    // issue next step's chunk-0 (x, read-only) loads BEFORE the barrier:
    // they complete during the barrier spin, off the critical path.
    if (t + 1 < T_) {
      const unsigned short* xnext = Xb + ((t + 1) * B_) * E_;
      #pragma unroll
      for (int j = 0; j < 4; ++j) aC[j] = ldA(xnext + xoffj[j]);
    }

    // grid-wide release/acquire barrier across the 16 persistent WGs
    // (its __syncthreads also protects zbuf reuse)
    __threadfence();
    __syncthreads();
    if (tid == 0) {
      __hip_atomic_fetch_add(barrier_cnt, 1u, __ATOMIC_RELEASE, __HIP_MEMORY_SCOPE_AGENT);
      const unsigned target = (unsigned)(t + 1) * NWG_REC;
      while (__hip_atomic_load(barrier_cnt, __ATOMIC_ACQUIRE, __HIP_MEMORY_SCOPE_AGENT) < target)
        __builtin_amdgcn_s_sleep(1);
    }
    __syncthreads();
    __threadfence();   // acquire side: make other WGPs' h stores visible
  }
}

// ---------------- kernel 4: y = relu(h @ Wd + bd), in-place on d_out ----------------
__global__ __launch_bounds__(256) void dense_relu(
    const unsigned short* __restrict__ WdT,   // [D][U] bf16 (B-operand layout)
    const float* __restrict__ bd,
    float* __restrict__ y) {                  // holds h on entry, y on exit
  __shared__ __align__(16) unsigned short Hs[16 * U_];   // 8 KB
  const int tid = threadIdx.x, lane = tid & 31, wave = tid >> 5;
  const int r0 = blockIdx.x * 16;
  const int half = lane >> 4;
  const int n16 = lane & 15;

  // stage this WG's 16 rows (read fully before any overwrite)
  for (int i = tid; i < 16 * U_; i += 256) {
    const int rr = i >> 8, k = i & 255;
    Hs[i] = f2bf(y[(r0 + rr) * U_ + k]);
  }
  __syncthreads();

  const int aoff = n16 * U_ + half * 8;               // A from LDS rows
  #pragma unroll
  for (int nt = 0; nt < 2; ++nt) {
    const int n = wave * 2 + nt;                      // n-tile 0..15
    const float bdv = bd[n * 16 + n16];
    const int boff = (n * 16 + n16) * U_ + half * 8;  // B from WdT
    v8f acc;
    #pragma unroll
    for (int k = 0; k < 8; ++k) acc[k] = bdv;

    #pragma unroll
    for (int kk = 0; kk < U_; kk += 32) {
      const v16bf a  = ldA(&Hs[aoff + kk]);
      const v16bf bm = ldB(WdT + boff + kk);
      acc = __builtin_amdgcn_wmma_f32_16x16x32_bf16(
          false, a, false, bm, (short)0, acc, false, false);
    }
    #pragma unroll
    for (int k = 0; k < 8; ++k) {
      float v = acc[k];
      v = v > 0.0f ? v : 0.0f;
      y[(r0 + k + 8 * half) * D_ + n * 16 + n16] = v;
    }
  }
}

// ---------------- launcher ----------------
extern "C" void kernel_launch(void* const* d_in, const int* in_sizes, int n_in,
                              void* d_out, int out_size, void* d_ws, size_t ws_size,
                              hipStream_t stream) {
  (void)in_sizes; (void)n_in; (void)out_size; (void)ws_size;
  const int*   tokens = (const int*)d_in[0];
  const float* emb    = (const float*)d_in[1];
  const float* Wk     = (const float*)d_in[2];
  const float* Wr     = (const float*)d_in[3];
  const float* bias   = (const float*)d_in[4];
  const float* Wd     = (const float*)d_in[5];
  const float* bd     = (const float*)d_in[6];
  float* y = (float*)d_out;

  char* ws = (char*)d_ws;
  size_t off = 0;
  auto alloc = [&](size_t bytes) -> void* {
    void* p = ws + off;
    off += (bytes + 255) & ~(size_t)255;
    return p;
  };
  unsigned short* Xb   = (unsigned short*)alloc((size_t)T_ * B_ * E_ * 2); // 32 MB
  unsigned short* WkT  = (unsigned short*)alloc((size_t)NG * U_ * E_ * 2);
  unsigned short* WrT  = (unsigned short*)alloc((size_t)NG * U_ * U_ * 2);
  unsigned short* WdT  = (unsigned short*)alloc((size_t)D_ * U_ * 2);
  float*          maskT = (float*)alloc((size_t)T_ * B_ * 4);
  unsigned short* hbuf = (unsigned short*)alloc((size_t)2 * B_ * U_ * 2);
  unsigned*       cnt  = (unsigned*)alloc(256);

  (void)hipMemsetAsync(hbuf, 0, (size_t)2 * B_ * U_ * 2, stream);  // h_0 = 0
  (void)hipMemsetAsync(cnt, 0, 256, stream);                       // barrier gen = 0

  gather_embed<<<T_ * B_, 128, 0, stream>>>(tokens, emb, Xb, maskT);
  const int prep_elems = NG * U_ * E_ + NG * U_ * U_ + D_ * U_;
  prep_weights<<<(prep_elems + 255) / 256, 256, 0, stream>>>(Wk, Wr, Wd, WkT, WrT, WdT);
  lstm_rec<<<NWG_REC, 256, 0, stream>>>(Xb, WkT, WrT, bias, maskT, hbuf, cnt, y);
  dense_relu<<<(B_ * T_) / 16, 256, 0, stream>>>(WdT, bd, y);
}
